// MultiHeadSelfAttention_82231443849923
// MI455X (gfx1250) — compile-verified, matrix operand order checked by reading, not run on hardware
//
#include <hip/hip_runtime.h>

// Problem constants
#define CDIM 512
#define SEQ  1024
#define NB   8
#define NH   8
#define HD   64
#define MTOT (NB * SEQ)   // 8192 total token rows
#define KCH  32           // GEMM K chunk (one WMMA K depth)

typedef __attribute__((ext_vector_type(16))) _Float16 v16h;
typedef __attribute__((ext_vector_type(8)))  _Float16 v8h;
typedef __attribute__((ext_vector_type(8)))  float    v8f;

// ---- CDNA5 async global->LDS copy (ASYNCcnt path, ISA 15.18.3 op 98) ----
__device__ __forceinline__ unsigned lds_off_of(const void* p) {
  // LDS aperture keeps the wave-relative LDS byte offset in addr[31:0].
  return (unsigned)(unsigned long long)(uintptr_t)p;
}
__device__ __forceinline__ void async_b128(unsigned lds_off, const void* gaddr) {
  asm volatile("global_load_async_to_lds_b128 %0, %1, off"
               :: "v"(lds_off), "v"((unsigned long long)(uintptr_t)gaddr)
               : "memory");
}
__device__ __forceinline__ void wait_async0() {
  asm volatile("s_wait_asynccnt 0x0" ::: "memory");
}

// Load one WMMA f16 A/B fragment for a 32-wide K chunk.
// Lanes 0-15 hold K {0..7,16..23}, lanes 16-31 hold K {8..15,24..31}.
__device__ __forceinline__ v16h load_frag(const _Float16* p, int hi) {
  v8h a = *(const v8h*)(p + (hi ? 8 : 0));
  v8h b = *(const v8h*)(p + (hi ? 24 : 16));
  v16h r;
#pragma unroll
  for (int i = 0; i < 8; ++i) { r[i] = a[i]; r[i + 8] = b[i]; }
  return r;
}

// x [N, C, H*W] f32  ->  xs [N, S, C] f16
__global__ void pack_x_kernel(const float* __restrict__ x, _Float16* __restrict__ xs) {
  int idx = blockIdx.x * blockDim.x + threadIdx.x;
  if (idx >= NB * CDIM * SEQ) return;
  int n   = idx / (CDIM * SEQ);
  int rem = idx % (CDIM * SEQ);
  int c   = rem / SEQ;
  int s   = rem % SEQ;
  xs[((size_t)n * SEQ + s) * CDIM + c] = (_Float16)x[idx];
}

// All four weight matrices f32 -> f16 (row-major [C_out, C_in]; these rows ARE
// the B-fragment columns for y = x @ W^T).
__global__ void pack_w_kernel(const float* __restrict__ a, const float* __restrict__ b,
                              const float* __restrict__ c, const float* __restrict__ d,
                              _Float16* __restrict__ A, _Float16* __restrict__ B,
                              _Float16* __restrict__ Cw, _Float16* __restrict__ D) {
  int i = blockIdx.x * blockDim.x + threadIdx.x;
  if (i >= CDIM * CDIM) return;
  A[i]  = (_Float16)a[i];
  B[i]  = (_Float16)b[i];
  Cw[i] = (_Float16)c[i];
  D[i]  = (_Float16)d[i];
}

// Y[M,512] = A[M,512](f16) @ W[512,512]^T(f16 row-major) + bias.
// mode 0: f16 row-major [M,512]   (Q, K)
// mode 1: f32 row-major [M,512]   (final output)
// mode 2: f16 head-transposed Vt[(n*NH+h)*HD + d][SEQ]   (V)
// Block 256 thr = 8 waves (4 row-groups x 2 col-groups); block tile 128x128;
// wave tile 32x64 (8 accumulators). K chunks of 32 double-buffered in LDS via
// async copies overlapped with WMMA compute.
__global__ void gemm_bt_kernel(const _Float16* __restrict__ A,
                               const _Float16* __restrict__ W,
                               const float* __restrict__ bias,
                               _Float16* __restrict__ out16,
                               float* __restrict__ out32,
                               int mode) {
  __shared__ __align__(16) _Float16 At[2][128 * KCH];
  __shared__ __align__(16) _Float16 Wt[2][128 * KCH];

  const int tid  = threadIdx.x;
  const int lane = tid & 31;
  const int wave = tid >> 5;
  const int lr   = lane & 15;
  const int hi   = lane >> 4;
  const int wr   = wave & 3;   // row group 0..3
  const int wc   = wave >> 2;  // col group 0..1
  const int mblk = blockIdx.x * 128;
  const int nblk = blockIdx.y * 128;

  // Cooperative copy coords: each thread moves 32B of A and 32B of W per chunk.
  const int crow = tid >> 1;           // 0..127
  const int chof = (tid & 1) * 16;     // halves within 32-half row
  const _Float16* gA = A + (size_t)(mblk + crow) * CDIM + chof;
  const _Float16* gW = W + (size_t)(nblk + crow) * CDIM + chof;
  unsigned ldsA[2], ldsW[2];
#pragma unroll
  for (int b = 0; b < 2; ++b) {
    ldsA[b] = lds_off_of(&At[b][crow * KCH + chof]);
    ldsW[b] = lds_off_of(&Wt[b][crow * KCH + chof]);
  }
  auto issue = [&](int b, int kc) {
    const _Float16* a = gA + kc;
    const _Float16* w = gW + kc;
    async_b128(ldsA[b],      a);
    async_b128(ldsA[b] + 16, a + 8);
    async_b128(ldsW[b],      w);
    async_b128(ldsW[b] + 16, w + 8);
  };

  v8f acc[2][4];
#pragma unroll
  for (int s = 0; s < 2; ++s)
#pragma unroll
    for (int c = 0; c < 4; ++c) acc[s][c] = v8f{0.f,0.f,0.f,0.f,0.f,0.f,0.f,0.f};

  issue(0, 0);
  int cur = 0;
  for (int t = 0; t < CDIM / KCH; ++t) {
    wait_async0();        // my chunk-t copies done
    __syncthreads();      // everyone's chunk-t data visible; other buffer free
    if (t < CDIM / KCH - 1) issue(1 - cur, (t + 1) * KCH);

    const _Float16* Ab = At[cur];
    const _Float16* Wb = Wt[cur];
    v16h af[2];
#pragma unroll
    for (int s = 0; s < 2; ++s)
      af[s] = load_frag(Ab + (size_t)(wr * 32 + s * 16 + lr) * KCH, hi);
#pragma unroll
    for (int c = 0; c < 4; ++c) {
      v16h bf = load_frag(Wb + (size_t)(wc * 64 + c * 16 + lr) * KCH, hi);
#pragma unroll
      for (int s = 0; s < 2; ++s)
        acc[s][c] = __builtin_amdgcn_wmma_f32_16x16x32_f16(false, af[s], false, bf,
                                                           (short)0, acc[s][c], false, false);
    }
    cur ^= 1;
  }

#pragma unroll
  for (int c = 0; c < 4; ++c) {
    const int col = nblk + wc * 64 + c * 16 + lr;
    const float bv = bias[col];
#pragma unroll
    for (int s = 0; s < 2; ++s) {
#pragma unroll
      for (int r = 0; r < 8; ++r) {
        const int row = mblk + wr * 32 + s * 16 + r + 8 * hi;
        float y = acc[s][c][r] + bv;
        if (mode == 0) {
          out16[(size_t)row * CDIM + col] = (_Float16)y;
        } else if (mode == 1) {
          out32[(size_t)row * CDIM + col] = y;
        } else {
          // Vt[(n*NH + h)*HD + d][SEQ]: h = col/64, d = col%64, n = row/SEQ, s = row%SEQ
          const int hh = col >> 6, dd = col & 63;
          const int nn = row >> 10, ss = row & 1023;
          out16[((size_t)(nn * NH + hh) * HD + dd) * SEQ + ss] = (_Float16)y;
        }
      }
    }
  }
}

// Flash attention per (n, h): block = 128 threads (4 waves), each wave one
// 16-row q tile. K tile (32 x 64, row-major) and Vt tile (64 x 32, d-major)
// double-buffered in LDS via async copies. All WMMA fragments are contiguous
// 128-bit LDS reads. O written in [N, h, S, d] flat order.
__global__ void attn_kernel(const _Float16* __restrict__ Q,
                            const _Float16* __restrict__ K,
                            const _Float16* __restrict__ Vt,
                            _Float16* __restrict__ O) {
  __shared__ __align__(16) _Float16 Kt[2][32 * 64];
  __shared__ __align__(16) _Float16 VtT[2][64 * 32];
  __shared__ __align__(16) _Float16 Pt[4][16 * 32];

  const int tid  = threadIdx.x;
  const int lane = tid & 31;
  const int wave = tid >> 5;
  const int lr   = lane & 15;
  const int hi   = lane >> 4;
  const int n    = blockIdx.y >> 3;   // NH == 8
  const int h    = blockIdx.y & 7;
  const int q0   = blockIdx.x * 64 + wave * 16;

  const _Float16* Qb  = Q  + (size_t)n * SEQ * CDIM + h * HD;
  const _Float16* Kb  = K  + (size_t)n * SEQ * CDIM + h * HD;
  const _Float16* VtB = Vt + (size_t)(n * NH + h) * HD * SEQ;   // [64][SEQ]

  const v16h qa0 = load_frag(Qb + (size_t)(q0 + lr) * CDIM, hi);
  const v16h qa1 = load_frag(Qb + (size_t)(q0 + lr) * CDIM + 32, hi);

  v8f o[4];
#pragma unroll
  for (int c = 0; c < 4; ++c) o[c] = v8f{0.f,0.f,0.f,0.f,0.f,0.f,0.f,0.f};
  float m[8], l[8];
#pragma unroll
  for (int r = 0; r < 8; ++r) { m[r] = -1.0e30f; l[r] = 0.0f; }

  // Cooperative copy coords.
  // K tile: 32 rows x 128B -> thread: row = tid/4, 32B at (tid%4)*32B.
  const int krow = tid >> 2;
  const int kcb  = (tid & 3) * 16;     // halves
  // Vt tile: 64 rows x 64B -> thread: row = tid/2, 32B at (tid%2)*32B.
  const int vrow = tid >> 1;
  const int vcb  = (tid & 1) * 16;     // halves
  const _Float16* gK = Kb  + (size_t)krow * CDIM + kcb;
  const _Float16* gV = VtB + (size_t)vrow * SEQ  + vcb;
  unsigned ldsK[2], ldsV[2];
#pragma unroll
  for (int b = 0; b < 2; ++b) {
    ldsK[b] = lds_off_of(&Kt[b][krow * 64 + kcb]);
    ldsV[b] = lds_off_of(&VtT[b][vrow * 32 + vcb]);
  }
  auto issueKV = [&](int b, int k0) {
    const _Float16* kk = gK + (size_t)k0 * CDIM;  // advance 32 k-rows
    const _Float16* vv = gV + k0;                 // advance 32 halves along S
    async_b128(ldsK[b],      kk);
    async_b128(ldsK[b] + 16, kk + 8);
    async_b128(ldsV[b],      vv);
    async_b128(ldsV[b] + 16, vv + 8);
  };

  issueKV(0, 0);
  int cur = 0;
  for (int kt = 0; kt < SEQ / 32; ++kt) {
    wait_async0();
    __syncthreads();
    if (kt < SEQ / 32 - 1) issueKV(1 - cur, (kt + 1) * 32);

    const _Float16* Kc = Kt[cur];
    const _Float16* Vc = VtT[cur];

    // S_tile (16 x 32) = q_tile @ k_tile^T  (K = 64 -> 2 chunks)
    v8f s0 = v8f{0.f,0.f,0.f,0.f,0.f,0.f,0.f,0.f};
    v8f s1 = v8f{0.f,0.f,0.f,0.f,0.f,0.f,0.f,0.f};
    {
      v16h b;
      b = load_frag(Kc + (size_t)lr * 64 + 0,  hi);
      s0 = __builtin_amdgcn_wmma_f32_16x16x32_f16(false, qa0, false, b, (short)0, s0, false, false);
      b = load_frag(Kc + (size_t)lr * 64 + 32, hi);
      s0 = __builtin_amdgcn_wmma_f32_16x16x32_f16(false, qa1, false, b, (short)0, s0, false, false);
      b = load_frag(Kc + (size_t)(16 + lr) * 64 + 0,  hi);
      s1 = __builtin_amdgcn_wmma_f32_16x16x32_f16(false, qa0, false, b, (short)0, s1, false, false);
      b = load_frag(Kc + (size_t)(16 + lr) * 64 + 32, hi);
      s1 = __builtin_amdgcn_wmma_f32_16x16x32_f16(false, qa1, false, b, (short)0, s1, false, false);
    }

    // Online softmax, row-wise; row (r + 8*hi) lives across a 16-lane group.
#pragma unroll
    for (int r = 0; r < 8; ++r) {
      float v0 = s0[r] * 0.125f;           // 1/sqrt(64)
      float v1 = s1[r] * 0.125f;
      float rm = fmaxf(v0, v1);
#pragma unroll
      for (int off = 1; off < 16; off <<= 1)
        rm = fmaxf(rm, __shfl_xor(rm, off, 32));
      float mn    = fmaxf(m[r], rm);
      float alpha = __expf(m[r] - mn);
      float p0    = __expf(v0 - mn);
      float p1    = __expf(v1 - mn);
      float rs    = p0 + p1;
#pragma unroll
      for (int off = 1; off < 16; off <<= 1)
        rs += __shfl_xor(rs, off, 32);
      l[r] = l[r] * alpha + rs;
      m[r] = mn;
#pragma unroll
      for (int c = 0; c < 4; ++c) o[c][r] *= alpha;
      int prow = r + 8 * hi;
      Pt[wave][prow * 32 + lr]      = (_Float16)p0;   // C-layout -> A-layout
      Pt[wave][prow * 32 + 16 + lr] = (_Float16)p1;   // via per-wave LDS region
    }

    // O += P(16x32) @ V_tile(32x64): B fragments are contiguous Vt rows.
    v16h pa = load_frag(&Pt[wave][(size_t)lr * 32], hi);
#pragma unroll
    for (int c = 0; c < 4; ++c) {
      v16h vb = load_frag(Vc + (size_t)(c * 16 + lr) * 32, hi);
      o[c] = __builtin_amdgcn_wmma_f32_16x16x32_f16(false, pa, false, vb,
                                                    (short)0, o[c], false, false);
    }
    cur ^= 1;
  }

  // Normalize and store in [N, h, S, d] flat order.
  _Float16* Ob = O + ((size_t)(n * NH + h) * SEQ + q0) * HD;
#pragma unroll
  for (int r = 0; r < 8; ++r) {
    float inv = 1.0f / l[r];
    int row = r + 8 * hi;
#pragma unroll
    for (int c = 0; c < 4; ++c)
      Ob[(size_t)row * HD + c * 16 + lr] = (_Float16)(o[c][r] * inv);
  }
}

extern "C" void kernel_launch(void* const* d_in, const int* in_sizes, int n_in,
                              void* d_out, int out_size, void* d_ws, size_t ws_size,
                              hipStream_t stream) {
  (void)in_sizes; (void)n_in; (void)out_size; (void)ws_size;
  const float* x  = (const float*)d_in[0];
  const float* Wq = (const float*)d_in[1];
  const float* bq = (const float*)d_in[2];
  const float* Wk = (const float*)d_in[3];
  const float* bk = (const float*)d_in[4];
  const float* Wv = (const float*)d_in[5];
  const float* bv = (const float*)d_in[6];
  const float* Wo = (const float*)d_in[7];
  const float* bo = (const float*)d_in[8];
  float* out = (float*)d_out;

  // Workspace layout (offsets in f16 elements). Total ~42 MB.
  _Float16* ws = (_Float16*)d_ws;
  const size_t NSC = (size_t)NB * SEQ * CDIM;   // 4,194,304
  const size_t WSZ = (size_t)CDIM * CDIM;       //   262,144
  _Float16* xs   = ws;
  _Float16* wq16 = xs + NSC;
  _Float16* wk16 = wq16 + WSZ;
  _Float16* wv16 = wk16 + WSZ;
  _Float16* wo16 = wv16 + WSZ;
  _Float16* Qh   = wo16 + WSZ;
  _Float16* Kh   = Qh + NSC;
  _Float16* Vth  = Kh + NSC;   // V stored head-transposed [(n*NH+h)*HD+d][SEQ]
  _Float16* Oh   = Vth + NSC;

  // 1) Pack activations and weights to f16
  pack_x_kernel<<<(NB * CDIM * SEQ + 255) / 256, 256, 0, stream>>>(x, xs);
  pack_w_kernel<<<(CDIM * CDIM + 255) / 256, 256, 0, stream>>>(
      Wq, Wk, Wv, Wo, wq16, wk16, wv16, wo16);

  // 2) Q/K/V projections (fused bias); V written head-transposed
  dim3 ggrid(MTOT / 128, CDIM / 128);
  gemm_bt_kernel<<<ggrid, 256, 0, stream>>>(xs, wq16, bq, Qh,  nullptr, 0);
  gemm_bt_kernel<<<ggrid, 256, 0, stream>>>(xs, wk16, bk, Kh,  nullptr, 0);
  gemm_bt_kernel<<<ggrid, 256, 0, stream>>>(xs, wv16, bv, Vth, nullptr, 2);

  // 3) Flash attention per (n, h); O stored [N, h, S, d] flat
  attn_kernel<<<dim3(SEQ / 64, NB * NH), 128, 0, stream>>>(Qh, Kh, Vth, Oh);

  // 4) Output projection: O (flat [8192, 512]) @ Wo^T + bo -> f32 d_out
  //    (both reference reshapes are memory-order-preserving)
  gemm_bt_kernel<<<ggrid, 256, 0, stream>>>(Oh, wo16, bo, nullptr, out, 1);
}
